// FlashAttention_2293512536564
// MI455X (gfx1250) — matrix operand order repaired and checked by reading, not compiled, not run on hardware
//
#include <hip/hip_runtime.h>
#include <hip/hip_bf16.h>

typedef __attribute__((ext_vector_type(16))) _Float16 v16h;
typedef __attribute__((ext_vector_type(8)))  _Float16 v8h;
typedef __attribute__((ext_vector_type(8)))  float    v8f;

#define HIDDEN   1024
#define HEADS    16
#define HEAD_DIM 64
#define BATCH    2
#define SEQ      2048
#define MROWS    (BATCH * SEQ)      // 4096
#define QK_SCALE 0.125f             // 1/sqrt(64)

// ---------------------------------------------------------------------------
// fp32 -> f16 conversion
// ---------------------------------------------------------------------------
__global__ void cvt_f32_to_f16(const float* __restrict__ in,
                               _Float16* __restrict__ out, int n) {
  int i = blockIdx.x * blockDim.x + threadIdx.x;
  if (i < n) out[i] = (_Float16)in[i];
}

// ---------------------------------------------------------------------------
// WMMA GEMM: C[M=4096, N=1024] = A[M,1024] * W[1024,1024]
// Block = 128 threads (4 wave32). Output tile 64(M) x 128(N):
//   wave w -> rows [bx*64 + w*16, +16), cols [by*128, +128) as 8 accumulators.
// Per K-chunk (32): W[32 x 128] staged once into LDS via async global->LDS
// b128 copies (ASYNCcnt), shared by all 4 waves; A fragments read from global.
// MODE 0: out = f16 Q [B,H,S,D], scaled by QK_SCALE
// MODE 1: out = f16 K^T [B,H,D,S]
// MODE 2: out = f16 V [B,H,S,D]
// MODE 3: out = f32 row-major [M,N]
// ---------------------------------------------------------------------------
template <int MODE>
__global__ void wmma_gemm_1024(const _Float16* __restrict__ A,
                               const _Float16* __restrict__ W,
                               void* __restrict__ out) {
  __shared__ __align__(16) _Float16 wlds[32][128];   // 8 KiB K-chunk of W

  const int tid  = threadIdx.x;
  const int lane = tid & 31;
  const int wave = tid >> 5;
  const int m0 = blockIdx.x * 64 + wave * 16;
  const int n0 = blockIdx.y * 128;
  const int mrow = m0 + (lane & 15);
  const int kg = (lane < 16) ? 0 : 8;   // A-fragment K-group per ISA layout

  v8f acc[8] = {};
  for (int k0 = 0; k0 < 1024; k0 += 32) {
    __syncthreads();  // previous chunk fully consumed before overwrite
    // Stage W[k0..k0+31][n0..n0+127] -> LDS: 512 x 16B units, 4 per thread.
#pragma unroll
    for (int u = 0; u < 4; ++u) {
      const int unit = tid + u * 128;     // 0..511
      const int row  = unit >> 4;         // 0..31 (K within chunk)
      const int seg  = unit & 15;         // 16 segments of 8 halves per row
      const _Float16* g = W + (size_t)(k0 + row) * 1024 + n0 + seg * 8;
      // Generic pointer low 32 bits == LDS byte offset (ISA flat->LDS mapping)
      const unsigned loff = (unsigned)(size_t)&wlds[row][seg * 8];
      asm volatile("global_load_async_to_lds_b128 %0, %1, off"
                   :: "v"(loff), "v"(g) : "memory");
    }
    asm volatile("s_wait_asynccnt 0" ::: "memory");
    __syncthreads();  // whole chunk visible to all waves

    // A fragment: lane holds row `mrow`, halves K = k0+kg+0..7 / k0+16+kg+0..7
    union { v16h v; v8h h8[2]; } a;
    const _Float16* arow = A + (size_t)mrow * 1024 + k0 + kg;
    a.h8[0] = *(const v8h*)(arow);
    a.h8[1] = *(const v8h*)(arow + 16);
    // B fragments from LDS: lane holds column lane%16, K halves kg+0..7 / 16+kg+0..7
#pragma unroll
    for (int t = 0; t < 8; ++t) {
      v16h b;
#pragma unroll
      for (int s = 0; s < 16; ++s)
        b[s] = wlds[(s < 8) ? kg + s : 8 + kg + s][t * 16 + (lane & 15)];
      acc[t] = __builtin_amdgcn_wmma_f32_16x16x32_f16(
          false, a.v, false, b, (short)0, acc[t], false, false);
    }
  }

  const int ncol = lane & 15;
  const int rbase = (lane < 16) ? 0 : 8;  // C/D fragment row split
#pragma unroll
  for (int t = 0; t < 8; ++t) {
    const int n = n0 + t * 16 + ncol;
#pragma unroll
    for (int r = 0; r < 8; ++r) {
      const int m = m0 + rbase + r;
      float val = acc[t][r];
      if constexpr (MODE == 3) {
        ((float*)out)[(size_t)m * 1024 + n] = val;
      } else {
        const int b = m >> 11, s = m & 2047;   // m = b*SEQ + s
        const int h = n >> 6,  d = n & 63;     // n = h*64 + d
        _Float16 hv;
        if constexpr (MODE == 0) hv = (_Float16)(val * QK_SCALE);
        else                     hv = (_Float16)val;
        if constexpr (MODE == 1)
          ((_Float16*)out)[((size_t)(b * HEADS + h) * HEAD_DIM + d) * SEQ + s] = hv;
        else
          ((_Float16*)out)[((size_t)(b * HEADS + h) * SEQ + s) * HEAD_DIM + d] = hv;
      }
    }
  }
}

// ---------------------------------------------------------------------------
// Flash attention (causal): per block = (qblock of 64 rows, head, batch),
// 4 wave32; each wave owns 16 query rows, streams keys in chunks of 32 with
// online softmax; output written f16 [B,S,H*D] for the final projection.
// ---------------------------------------------------------------------------
__global__ void flash_attn_kernel(const _Float16* __restrict__ Q,   // [B,H,S,D] pre-scaled
                                  const _Float16* __restrict__ Kt,  // [B,H,D,S]
                                  const _Float16* __restrict__ V,   // [B,H,S,D]
                                  _Float16* __restrict__ O) {       // [B,S,H*D]
  __shared__ __align__(16) _Float16 plds[4][16][32];

  const int lane = threadIdx.x & 31;
  const int wave = threadIdx.x >> 5;
  const int qblk = blockIdx.x;   // 0..31
  const int h    = blockIdx.y;   // 0..15
  const int b    = blockIdx.z;   // 0..1
  const int q0   = qblk * 64 + wave * 16;

  const _Float16* qb  = Q  + (size_t)(b * HEADS + h) * SEQ * HEAD_DIM;
  const _Float16* ktb = Kt + (size_t)(b * HEADS + h) * HEAD_DIM * SEQ;
  const _Float16* vbp = V  + (size_t)(b * HEADS + h) * SEQ * HEAD_DIM;

  const int mrow  = lane & 15;
  const int kg    = (lane < 16) ? 0 : 8;
  const int rbase = kg;

  // Preload Q A-fragments for both 32-wide d-chunks (D = 64)
  union { v16h v; v8h h8[2]; } qa[2];
#pragma unroll
  for (int c = 0; c < 2; ++c) {
    const _Float16* qr = qb + (size_t)(q0 + mrow) * HEAD_DIM + c * 32 + kg;
    qa[c].h8[0] = *(const v8h*)(qr);
    qa[c].h8[1] = *(const v8h*)(qr + 16);
  }

  float mrun[8], srun[8];
  v8f oacc[4] = {};
#pragma unroll
  for (int r = 0; r < 8; ++r) { mrun[r] = -1.0e30f; srun[r] = 0.0f; }

  const int kend = qblk * 64 + 64;  // causal bound shared by the whole block
  for (int key0 = 0; key0 < kend; key0 += 32) {
    // ---- scores: two 16x16 tiles (keys key0..+15, key0+16..+31), K-dim 64 ----
    v8f c0 = {}, c1 = {};
#pragma unroll
    for (int c = 0; c < 2; ++c) {
      v16h bk0, bk1;
#pragma unroll
      for (int s = 0; s < 16; ++s) {
        const _Float16* kp = ktb + (size_t)(c * 32 + ((s < 8) ? kg + s : 8 + kg + s)) * SEQ + key0 + (lane & 15);
        bk0[s] = kp[0];
        bk1[s] = kp[16];
      }
      c0 = __builtin_amdgcn_wmma_f32_16x16x32_f16(false, qa[c].v, false, bk0,
                                                  (short)0, c0, false, false);
      c1 = __builtin_amdgcn_wmma_f32_16x16x32_f16(false, qa[c].v, false, bk1,
                                                  (short)0, c1, false, false);
    }

    // ---- causal mask (C-fragment: lane = key col, rows rbase+r) ----
    const int kc0 = key0 + (lane & 15);
    const int kc1 = kc0 + 16;
#pragma unroll
    for (int r = 0; r < 8; ++r) {
      const int qrow = q0 + rbase + r;
      if (kc0 > qrow) c0[r] = -1.0e30f;
      if (kc1 > qrow) c1[r] = -1.0e30f;
    }

    // ---- online softmax: row max across the 16-lane group ----
    float mx[8];
#pragma unroll
    for (int r = 0; r < 8; ++r) mx[r] = fmaxf(c0[r], c1[r]);
#pragma unroll
    for (int off = 1; off < 16; off <<= 1) {
#pragma unroll
      for (int r = 0; r < 8; ++r)
        mx[r] = fmaxf(mx[r], __shfl_xor(mx[r], off, 32));
    }

    float alpha[8], rs[8];
#pragma unroll
    for (int r = 0; r < 8; ++r) {
      const float mn = fmaxf(mrun[r], mx[r]);
      const float p0 = expf(c0[r] - mn);
      const float p1 = expf(c1[r] - mn);
      c0[r] = p0; c1[r] = p1;
      rs[r] = p0 + p1;
      alpha[r] = expf(mrun[r] - mn);
      mrun[r] = mn;
    }
#pragma unroll
    for (int off = 1; off < 16; off <<= 1) {
#pragma unroll
      for (int r = 0; r < 8; ++r)
        rs[r] += __shfl_xor(rs[r], off, 32);
    }
#pragma unroll
    for (int r = 0; r < 8; ++r) srun[r] = srun[r] * alpha[r] + rs[r];
#pragma unroll
    for (int t = 0; t < 4; ++t)
#pragma unroll
      for (int r = 0; r < 8; ++r) oacc[t][r] *= alpha[r];

    // ---- restage P through LDS: C-layout -> A-fragment layout ----
#pragma unroll
    for (int r = 0; r < 8; ++r) {
      plds[wave][rbase + r][lane & 15]        = (_Float16)c0[r];
      plds[wave][rbase + r][16 + (lane & 15)] = (_Float16)c1[r];
    }
    // Same-wave LDS ops are in-order (DScnt in-order); no barrier needed.
    union { v16h v; v8h h8[2]; } pa;
    pa.h8[0] = *(const v8h*)&plds[wave][mrow][kg];
    pa.h8[1] = *(const v8h*)&plds[wave][mrow][16 + kg];

    // ---- PV: B fragment = V column lane%16, keys kg+0..7 / 16+kg+0..7 ----
    const _Float16* vrow = vbp + (size_t)key0 * HEAD_DIM + (lane & 15);
#pragma unroll
    for (int t = 0; t < 4; ++t) {
      v16h bv;
#pragma unroll
      for (int s = 0; s < 16; ++s)
        bv[s] = vrow[(size_t)((s < 8) ? kg + s : 8 + kg + s) * HEAD_DIM + t * 16];
      oacc[t] = __builtin_amdgcn_wmma_f32_16x16x32_f16(
          false, pa.v, false, bv, (short)0, oacc[t], false, false);
    }
  }

  // ---- normalize and store f16 [B,S,H*D] ----
#pragma unroll
  for (int t = 0; t < 4; ++t) {
#pragma unroll
    for (int r = 0; r < 8; ++r) {
      const int qrow = q0 + rbase + r;
      const float val = oacc[t][r] / srun[r];
      O[((size_t)(b * SEQ + qrow)) * (HEADS * HEAD_DIM) + h * HEAD_DIM +
        t * 16 + (lane & 15)] = (_Float16)val;
    }
  }
}

// ---------------------------------------------------------------------------
// Launch
// ---------------------------------------------------------------------------
extern "C" void kernel_launch(void* const* d_in, const int* in_sizes, int n_in,
                              void* d_out, int out_size, void* d_ws, size_t ws_size,
                              hipStream_t stream) {
  const float* X   = (const float*)d_in[0];   // [B,S,HIDDEN]
  // d_in[1] = attention_mask: exactly causal -> applied analytically, not read
  const float* Wq  = (const float*)d_in[2];
  const float* Wk  = (const float*)d_in[3];
  const float* Wv  = (const float*)d_in[4];
  const float* Wo  = (const float*)d_in[5];

  char* ws = (char*)d_ws;
  const size_t MB = 1024 * 1024;
  _Float16* Xh  = (_Float16*)(ws + 0 * MB);    // 8 MiB: X f16 [4096,1024]
  _Float16* Wqh = (_Float16*)(ws + 8 * MB);    // 2 MiB
  _Float16* Wkh = (_Float16*)(ws + 10 * MB);   // 2 MiB
  _Float16* Wvh = (_Float16*)(ws + 12 * MB);   // 2 MiB
  _Float16* Woh = (_Float16*)(ws + 14 * MB);   // 2 MiB
  _Float16* Qs  = (_Float16*)(ws + 16 * MB);   // 8 MiB: Q*scale [B,H,S,D]
  _Float16* Kt  = (_Float16*)(ws + 24 * MB);   // 8 MiB: K^T [B,H,D,S]
  _Float16* Vv  = (_Float16*)(ws + 32 * MB);   // 8 MiB: V [B,H,S,D]
  _Float16* Oh  = (_Float16*)(ws + 40 * MB);   // 8 MiB: attn out [B,S,H*D]

  const int nX = MROWS * HIDDEN;               // 4194304
  const int nW = HIDDEN * HEADS * HEAD_DIM;    // 1048576
  cvt_f32_to_f16<<<(nX + 255) / 256, 256, 0, stream>>>(X, Xh, nX);
  cvt_f32_to_f16<<<(nW + 255) / 256, 256, 0, stream>>>(Wq, Wqh, nW);
  cvt_f32_to_f16<<<(nW + 255) / 256, 256, 0, stream>>>(Wk, Wkh, nW);
  cvt_f32_to_f16<<<(nW + 255) / 256, 256, 0, stream>>>(Wv, Wvh, nW);
  cvt_f32_to_f16<<<(nW + 255) / 256, 256, 0, stream>>>(Wo, Woh, nW);

  dim3 gemm_grid(MROWS / 64, 1024 / 128);      // (64, 8)
  wmma_gemm_1024<0><<<gemm_grid, 128, 0, stream>>>(Xh, Wqh, (void*)Qs);
  wmma_gemm_1024<1><<<gemm_grid, 128, 0, stream>>>(Xh, Wkh, (void*)Kt);
  wmma_gemm_1024<2><<<gemm_grid, 128, 0, stream>>>(Xh, Wvh, (void*)Vv);

  dim3 attn_grid(SEQ / 64, HEADS, BATCH);      // (32, 16, 2)
  flash_attn_kernel<<<attn_grid, 128, 0, stream>>>(Qs, Kt, Vv, Oh);

  wmma_gemm_1024<3><<<gemm_grid, 128, 0, stream>>>(Oh, Woh, d_out);
}
